// Coordinator_12472585027727
// MI455X (gfx1250) — matrix operand order repaired and checked by reading, not compiled, not run on hardware
//
#include <hip/hip_runtime.h>
#include <hip/hip_bf16.h>

// ---------------- problem constants ----------------
#define NA 8       // agents
#define NDM 16     // ND (rows incl dummy)
#define NB 1024    // batch
#define PP 128     // plan
#define HH 256     // hidden

typedef __attribute__((ext_vector_type(16))) __bf16 v16bf;
typedef __attribute__((ext_vector_type(8)))  float  v8f;

union ABf {
  v16bf v;
  unsigned short us[16];
  uint4 q[2];
};

__device__ __forceinline__ unsigned short f2bf(float f) {
  unsigned int u = __float_as_uint(f);
  u += 0x7FFFu + ((u >> 16) & 1u);
  return (unsigned short)(u >> 16);
}
__device__ __forceinline__ float bf2f(unsigned short s) {
  return __uint_as_float(((unsigned int)s) << 16);
}

// ---------------- workspace layout (bytes) ----------------
static const size_t SZ_W3H = (size_t)NA * 3 * HH * (2 * PP) * 2;   // 8*768*256 bf16
static const size_t O_SEQ   = 0;                                   // int[1024]
static const size_t O_EMPTY = O_SEQ   + 4096;                      // int[1024]
static const size_t O_GPACK = O_EMPTY + 4096;                      // int[16384]
static const size_t O_GSCAT = O_GPACK + 65536;                     // int[16384]
static const size_t O_REV   = O_GSCAT + 65536;                     // int[16384]
static const size_t O_WIF   = O_REV   + 65536;                     // bf16 N-major (8,768,256)
static const size_t O_WHF   = O_WIF + SZ_W3H;
static const size_t O_WIB   = O_WHF + SZ_W3H;
static const size_t O_WHB   = O_WIB + SZ_W3H;
static const size_t O_W1T   = O_WHB + SZ_W3H;                      // bf16 (8,256,512)
static const size_t O_CSEQ  = O_W1T + (size_t)NA * 512 * 256 * 2;  // bf16 (8,16,1024,256)
static const size_t O_HA    = O_CSEQ + (size_t)NA * NDM * NB * HH * 2;
static const size_t O_HB    = O_HA + (size_t)16 * NB * HH * 4;     // f32 (16,1024,256)
static const size_t O_OUT   = O_HB + (size_t)16 * NB * HH * 4;     // bf16 (16,16,1024,256)
static const size_t O_HBFA  = O_OUT + (size_t)16 * NDM * NB * HH * 2;
static const size_t O_HBFB  = O_HBFA + (size_t)16 * NB * HH * 2;   // bf16 shadows of h

// =====================================================================
// prep: masks, seq_len, flat-order pack/scatter indices (torch semantics)
// =====================================================================
__global__ void __launch_bounds__(1024) prep_kernel(
    const float* __restrict__ plans, const float* __restrict__ commp,
    int* __restrict__ seq, int* __restrict__ emptyf,
    int* __restrict__ gpack, int* __restrict__ gscat, int* __restrict__ rev)
{
  __shared__ unsigned short s_mb[1024];
  __shared__ unsigned short s_sl[1024];
  __shared__ unsigned short s_src[16384];
  __shared__ int s_sm[1024];
  __shared__ int s_sp[1024];
  int b = threadIdx.x;
  {
    bool emp = true;
    for (int n = 0; n < NA && emp; ++n) {
      const float* r = commp + ((size_t)b * NA + n) * PP;
      for (int p = 0; p < PP; ++p) if (r[p] != 0.0f) { emp = false; break; }
    }
    unsigned int mb = 0xFF00u;  // dummy rows t=8..15 always valid
    int cnt = 0;
    for (int n = 0; n < NA; ++n) {
      const float* r = emp ? (plans + ((size_t)b * NA + n) * PP)
                           : (commp + ((size_t)b * NA + n) * PP);
      bool any = false;
      for (int p = 0; p < PP; ++p) if (r[p] != 0.0f) { any = true; break; }
      if (any) { mb |= (1u << n); cnt++; }
    }
    int sl = cnt + (NDM - NA);
    s_mb[b] = (unsigned short)mb;
    s_sl[b] = (unsigned short)sl;
    seq[b] = sl;
    emptyf[b] = emp ? 1 : 0;
  }
  __syncthreads();
  // per-thread chunk of 16 flat (t-major) elements
  int t = b >> 6;
  int b0 = (b & 63) * 16;
  int sm = 0, sp = 0;
  for (int e = 0; e < 16; ++e) {
    int bb = b0 + e;
    sm += (s_mb[bb] >> t) & 1;
    sp += (t < (int)s_sl[bb]) ? 1 : 0;
  }
  s_sm[b] = sm; s_sp[b] = sp;
  __syncthreads();
  for (int off = 1; off < 1024; off <<= 1) {  // inclusive Hillis-Steele scan
    int a0 = s_sm[b], a1 = s_sp[b];
    int c0 = 0, c1 = 0;
    if (b >= off) { c0 = s_sm[b - off]; c1 = s_sp[b - off]; }
    __syncthreads();
    s_sm[b] = a0 + c0; s_sp[b] = a1 + c1;
    __syncthreads();
  }
  int cm = s_sm[b] - sm;  // exclusive prefixes at chunk start
  int cp = s_sp[b] - sp;
  for (int e = 0; e < 16; ++e) {
    int bb = b0 + e;
    int f = t * NB + bb;
    int m = (s_mb[bb] >> t) & 1;
    int p = (t < (int)s_sl[bb]) ? 1 : 0;
    gscat[f] = m ? cm : -1;                          // cum_scat
    rev[f]   = p ? ((int)s_sl[bb] - 1 - t) : -1;     // rev_idx
    gpack[f] = p ? cp : -1;                          // temp: rank_pack
    if (m) { s_src[cm] = (unsigned short)f; cm++; }  // src_idx
    if (p) cp++;
  }
  __syncthreads();
  for (int e = 0; e < 16; ++e) {
    int f = t * NB + b0 + e;
    int r = gpack[f];
    gpack[f] = (r >= 0) ? (int)s_src[r] : -1;        // src_idx[rank_pack]
  }
}

// =====================================================================
// weight convert: f32 [a][rows][cols] -> bf16 N-major [a][rows][cols]
// (B-fragment friendly: lane = N column, contiguous K)
// =====================================================================
__global__ void cvtT_kernel(const float* __restrict__ src,
                            unsigned short* __restrict__ dst,
                            int rows, int cols)
{
  size_t total = (size_t)NA * rows * cols;
  size_t stride = (size_t)gridDim.x * blockDim.x;
  for (size_t i = (size_t)blockIdx.x * blockDim.x + threadIdx.x; i < total; i += stride) {
    int c = (int)(i % cols);
    size_t rr = i / cols;
    int r = (int)(rr % rows);
    int a = (int)(rr / rows);
    dst[((size_t)a * rows + r) * cols + c] = f2bf(src[i]);
  }
}

// plain f32 -> bf16 elementwise
__global__ void cvt_kernel(const float* __restrict__ src,
                           unsigned short* __restrict__ dst, size_t n)
{
  size_t stride = (size_t)gridDim.x * blockDim.x;
  for (size_t i = (size_t)blockIdx.x * blockDim.x + threadIdx.x; i < n; i += stride)
    dst[i] = f2bf(src[i]);
}

// =====================================================================
// build packed sequence (bf16), zeros past seq end
// =====================================================================
__global__ void build_cseq_kernel(
    const float* __restrict__ plans, const float* __restrict__ commp,
    const float* __restrict__ dummy, const int* __restrict__ gpack,
    const int* __restrict__ emptyf, unsigned short* __restrict__ cseq)
{
  int g = blockIdx.x * blockDim.x + threadIdx.x;
  int row = g >> 6;
  int c0 = (g & 63) * 4;
  int b = row & (NB - 1);
  int t = (row >> 10) & (NDM - 1);
  int agent = row >> 14;
  int gp = gpack[t * NB + b];
  unsigned short o0 = 0, o1 = 0, o2 = 0, o3 = 0;
  if (gp >= 0) {
    int st = gp >> 10, sb = gp & (NB - 1);
    float v[4];
    for (int q = 0; q < 4; ++q) {
      int c = c0 + q;
      if (st < NA) {
        if (c < PP) v[q] = plans[((size_t)sb * NA + agent) * PP + c];
        else {
          int cc = c - PP;
          v[q] = emptyf[sb] ? plans[((size_t)sb * NA + st) * PP + cc]
                            : commp[((size_t)sb * NA + st) * PP + cc];
        }
      } else {
        v[q] = dummy[(((size_t)agent * (NDM - NA) + (st - NA)) * NB + sb) * (2 * PP) + c];
      }
    }
    o0 = f2bf(v[0]); o1 = f2bf(v[1]); o2 = f2bf(v[2]); o3 = f2bf(v[3]);
  }
  *(ushort4*)(cseq + ((size_t)(agent * NDM + t) * NB + b) * (2 * PP) + c0) =
      make_ushort4(o0, o1, o2, o3);
}

// =====================================================================
// one GRU time step, fused gi+gh WMMA + gates, both directions
// one wave = 16(batch) x 16(hdim) tile; grid: agent*2048 + dir*1024 + rb*16 + hb
// =====================================================================
__device__ __forceinline__ v8f wmma_bf16(const ABf& a, const ABf& bb, v8f c) {
  return __builtin_amdgcn_wmma_f32_16x16x32_bf16(false, a.v, false, bb.v,
                                                 (short)0, c, false, false);
}

__global__ void __launch_bounds__(32) gru_step_kernel(
    const unsigned short* __restrict__ cseq,
    const unsigned short* __restrict__ wifT, const unsigned short* __restrict__ whfT,
    const unsigned short* __restrict__ wibT, const unsigned short* __restrict__ whbT,
    const float* __restrict__ hin, float* __restrict__ hout,
    const unsigned short* __restrict__ hbin, unsigned short* __restrict__ hbout,
    unsigned short* __restrict__ outbuf,
    const int* __restrict__ seq, const int* __restrict__ rev, int t)
{
  int gid = blockIdx.x;
  int hb = gid & 15;
  int rb = (gid >> 4) & 63;
  int dir = (gid >> 10) & 1;
  int agent = gid >> 11;
  int lane = threadIdx.x;
  int ll = lane & 15;
  int hi = lane >> 4;
  int ad = agent * 2 + dir;

  // A-matrix row for this lane (packed batch row); backward gathers reversed t
  int arow = rb * 16 + ll;
  int src_t = t;
  if (dir) {
    int rr = rev[t * NB + arow];
    src_t = (rr >= 0) ? rr : 0;  // garbage rows masked in epilogue
  }
  const unsigned short* xrow = cseq + ((size_t)(agent * NDM + src_t) * NB + arow) * (2 * PP);
  const unsigned short* hbrow = hbin + ((size_t)ad * NB + arow) * HH;

  const unsigned short* wiT = (dir ? wibT : wifT) + (size_t)agent * 256 * 768;
  const unsigned short* whT = (dir ? whbT : whfT) + (size_t)agent * 256 * 768;
  int nb0 = hb * 16;

  v8f accR = {}, accZ = {}, accNi = {}, accNh = {};
#pragma unroll
  for (int k = 0; k < 8; ++k) {
    int koff = k * 32;
    ABf ax, ah, bw;
    ax.q[0] = *(const uint4*)(xrow + koff + 8 * hi);
    ax.q[1] = *(const uint4*)(xrow + koff + 16 + 8 * hi);
    ah.q[0] = *(const uint4*)(hbrow + koff + 8 * hi);
    ah.q[1] = *(const uint4*)(hbrow + koff + 16 + 8 * hi);
    int kl = nb0 + ll;  // B fragment: lane = N column, K in table order
    const unsigned short* wi_k = wiT + (size_t)kl * 256 + koff + 8 * hi;
    const unsigned short* wh_k = whT + (size_t)kl * 256 + koff + 8 * hi;
    // r gate: gi + gh
    bw.q[0] = *(const uint4*)(wi_k);       bw.q[1] = *(const uint4*)(wi_k + 16);
    accR = wmma_bf16(ax, bw, accR);
    bw.q[0] = *(const uint4*)(wh_k);       bw.q[1] = *(const uint4*)(wh_k + 16);
    accR = wmma_bf16(ah, bw, accR);
    // z gate: gi + gh
    bw.q[0] = *(const uint4*)(wi_k + 65536); bw.q[1] = *(const uint4*)(wi_k + 65552);
    accZ = wmma_bf16(ax, bw, accZ);
    bw.q[0] = *(const uint4*)(wh_k + 65536); bw.q[1] = *(const uint4*)(wh_k + 65552);
    accZ = wmma_bf16(ah, bw, accZ);
    // n gate: keep gi and gh separate (n = tanh(gi_n + r*gh_n))
    bw.q[0] = *(const uint4*)(wi_k + 131072); bw.q[1] = *(const uint4*)(wi_k + 131088);
    accNi = wmma_bf16(ax, bw, accNi);
    bw.q[0] = *(const uint4*)(wh_k + 131072); bw.q[1] = *(const uint4*)(wh_k + 131088);
    accNh = wmma_bf16(ah, bw, accNh);
  }

  int col = hb * 16 + ll;
#pragma unroll
  for (int v = 0; v < 8; ++v) {
    int m = rb * 16 + v + 8 * hi;  // D layout: lane=N col, vgpr v -> M row v+8*hi
    float hp = hin[((size_t)ad * NB + m) * HH + col];
    float r = 1.0f / (1.0f + __expf(-accR[v]));
    float z = 1.0f / (1.0f + __expf(-accZ[v]));
    float n = tanhf(accNi[v] + r * accNh[v]);
    float hn = (1.0f - z) * n + z * hp;
    bool valid = (t < seq[m]);
    float hw = valid ? hn : hp;
    size_t hidx = ((size_t)ad * NB + m) * HH + col;
    hout[hidx] = hw;
    hbout[hidx] = f2bf(hw);  // bf16 shadow for next step's A-fragments
    outbuf[(((size_t)ad * NDM + t) * NB + m) * HH + col] = f2bf(valid ? hn : 0.0f);
  }
}

// =====================================================================
// scatter + un-reverse backward + LayerNorm + W1 (WMMA) + ReLU + W2
// one wave = 16 output rows of one agent
// =====================================================================
__global__ void __launch_bounds__(32) head_kernel(
    const unsigned short* __restrict__ outbuf,
    const int* __restrict__ gscat, const int* __restrict__ rev,
    const unsigned short* __restrict__ w1T,
    const float* __restrict__ lng, const float* __restrict__ lnb,
    const float* __restrict__ b1, const float* __restrict__ w2,
    const float* __restrict__ b2, float* __restrict__ omask)
{
  __shared__ __align__(32) unsigned char smem[49152];
  float (*s_pre)[512] = (float(*)[512])smem;                         // phase 1
  unsigned short (*s_a)[512] = (unsigned short(*)[512])(smem + 32768);
  float (*s_h1)[256] = (float(*)[256])smem;                          // reuses s_pre

  int agent = blockIdx.x >> 10;
  int blk = blockIdx.x & 1023;
  int lane = threadIdx.x;
  int jr = lane >> 1;
  int half = lane & 1;
  int j = blk * 16 + jr;  // flat (t*B+b) within agent
  int gs = gscat[j];
  float sum = 0.f, sq = 0.f;
  if (gs >= 0) {
    int st = gs >> 10, sb = gs & 1023;
    const unsigned short* fr = outbuf + (((size_t)(agent * 2 + 0) * NDM + st) * NB + sb) * HH;
    int bt = rev[st * NB + sb];
    const unsigned short* br = (bt >= 0)
        ? outbuf + (((size_t)(agent * 2 + 1) * NDM + bt) * NB + sb) * HH : (const unsigned short*)0;
    for (int e = 0; e < 256; ++e) {
      int c = half * 256 + e;
      float v = (c < HH) ? bf2f(fr[c]) : (br ? bf2f(br[c - HH]) : 0.0f);
      s_pre[jr][c] = v;
      sum += v; sq += v * v;
    }
  } else {
    for (int e = 0; e < 256; ++e) s_pre[jr][half * 256 + e] = 0.f;
  }
  float osum = __shfl_xor(sum, 1, 32);
  float osq  = __shfl_xor(sq, 1, 32);
  float mu = (sum + osum) * (1.0f / 512.0f);
  float var = (sq + osq) * (1.0f / 512.0f) - mu * mu;
  float rs = rsqrtf(var + 1e-5f);
  for (int e = 0; e < 256; ++e) {
    int c = half * 256 + e;
    float v = (s_pre[jr][c] - mu) * rs * lng[agent * 512 + c] + lnb[agent * 512 + c];
    s_a[jr][c] = f2bf(v);
  }
  __syncthreads();

  int ll = lane & 15, hi = lane >> 4;
  const unsigned short* arow = s_a[ll];
  for (int cb = 0; cb < 16; ++cb) {
    v8f acc = {};
#pragma unroll
    for (int k = 0; k < 16; ++k) {
      int koff = k * 32;
      ABf a, w;
      a.q[0] = *(const uint4*)(arow + koff + 8 * hi);
      a.q[1] = *(const uint4*)(arow + koff + 16 + 8 * hi);
      const unsigned short* wp = w1T + ((size_t)agent * 256 + cb * 16 + ll) * 512 + koff + 8 * hi;
      w.q[0] = *(const uint4*)(wp);
      w.q[1] = *(const uint4*)(wp + 16);
      acc = wmma_bf16(a, w, acc);
    }
    int col = cb * 16 + ll;
    float bias = b1[agent * HH + col];
#pragma unroll
    for (int v = 0; v < 8; ++v) {
      float hv = acc[v] + bias;
      s_h1[v + 8 * hi][col] = hv > 0.f ? hv : 0.f;
    }
  }
  __syncthreads();
  {
    int o = lane & 1;
    int row = lane >> 1;
    float s = b2[agent * 2 + o];
    const float* w2r = w2 + (size_t)(agent * 2 + o) * HH;
    for (int h = 0; h < HH; ++h) s += s_h1[row][h] * w2r[h];
    omask[(size_t)agent * (NDM * NB * 2) + (size_t)(blk * 16 + row) * 2 + o] = s;
  }
}

// =====================================================================
extern "C" void kernel_launch(void* const* d_in, const int* in_sizes, int n_in,
                              void* d_out, int out_size, void* d_ws, size_t ws_size,
                              hipStream_t stream) {
  (void)in_sizes; (void)n_in; (void)out_size; (void)ws_size;
  const float* plans = (const float*)d_in[0];
  const float* commp = (const float*)d_in[1];
  const float* hxs0  = (const float*)d_in[2];
  const float* dummy = (const float*)d_in[3];
  const float* Wi_f  = (const float*)d_in[4];
  const float* Wh_f  = (const float*)d_in[5];
  const float* Wi_b  = (const float*)d_in[6];
  const float* Wh_b  = (const float*)d_in[7];
  const float* lng   = (const float*)d_in[8];
  const float* lnb   = (const float*)d_in[9];
  const float* W1    = (const float*)d_in[10];
  const float* b1    = (const float*)d_in[11];
  const float* W2    = (const float*)d_in[12];
  const float* b2    = (const float*)d_in[13];

  char* ws = (char*)d_ws;
  int* seq    = (int*)(ws + O_SEQ);
  int* emptyf = (int*)(ws + O_EMPTY);
  int* gpack  = (int*)(ws + O_GPACK);
  int* gscat  = (int*)(ws + O_GSCAT);
  int* rev    = (int*)(ws + O_REV);
  unsigned short* wifT = (unsigned short*)(ws + O_WIF);
  unsigned short* whfT = (unsigned short*)(ws + O_WHF);
  unsigned short* wibT = (unsigned short*)(ws + O_WIB);
  unsigned short* whbT = (unsigned short*)(ws + O_WHB);
  unsigned short* w1T  = (unsigned short*)(ws + O_W1T);
  unsigned short* cseq = (unsigned short*)(ws + O_CSEQ);
  float* hA = (float*)(ws + O_HA);
  float* hB = (float*)(ws + O_HB);
  unsigned short* outbuf = (unsigned short*)(ws + O_OUT);
  unsigned short* hbfA = (unsigned short*)(ws + O_HBFA);
  unsigned short* hbfB = (unsigned short*)(ws + O_HBFB);

  prep_kernel<<<1, 1024, 0, stream>>>(plans, commp, seq, emptyf, gpack, gscat, rev);

  {
    int n3h = NA * 3 * HH * (2 * PP);
    int blocks = (n3h + 255) / 256;
    cvtT_kernel<<<blocks, 256, 0, stream>>>(Wi_f, wifT, 3 * HH, 2 * PP);
    cvtT_kernel<<<blocks, 256, 0, stream>>>(Wh_f, whfT, 3 * HH, HH);
    cvtT_kernel<<<blocks, 256, 0, stream>>>(Wi_b, wibT, 3 * HH, 2 * PP);
    cvtT_kernel<<<blocks, 256, 0, stream>>>(Wh_b, whbT, 3 * HH, HH);
    int n1 = NA * HH * (2 * HH);
    cvtT_kernel<<<(n1 + 255) / 256, 256, 0, stream>>>(W1, w1T, HH, 2 * HH);
  }

  // initial hiddens (N,2,B,H) -> hA (f32, identical layout) + bf16 shadow
  hipMemcpyAsync(hA, hxs0, (size_t)NA * 2 * NB * HH * 4, hipMemcpyDeviceToDevice, stream);
  {
    size_t nh = (size_t)NA * 2 * NB * HH;
    cvt_kernel<<<(int)((nh + 255) / 256), 256, 0, stream>>>(hxs0, hbfA, nh);
  }

  build_cseq_kernel<<<32768, 256, 0, stream>>>(plans, commp, dummy, gpack, emptyf, cseq);

  for (int t = 0; t < NDM; ++t) {
    const float*          hr  = (t & 1) ? hB : hA;
    float*                hw  = (t & 1) ? hA : hB;
    const unsigned short* hbr = (t & 1) ? hbfB : hbfA;
    unsigned short*       hbw = (t & 1) ? hbfA : hbfB;
    gru_step_kernel<<<NA * 2 * 64 * 16, 32, 0, stream>>>(
        cseq, wifT, whfT, wibT, whbT, hr, hw, hbr, hbw, outbuf, seq, rev, t);
  }
  // after t=15 (odd), final state lives in hA

  head_kernel<<<NA * 1024, 32, 0, stream>>>(outbuf, gscat, rev, w1T, lng, lnb,
                                            b1, W2, b2, (float*)d_out);

  hipMemcpyAsync((float*)d_out + (size_t)NA * NDM * NB * 2, hA,
                 (size_t)NA * 2 * NB * HH * 4, hipMemcpyDeviceToDevice, stream);
}
